// BitchyNetwork_39402029973866
// MI455X (gfx1250) — matrix operand order repaired and checked by reading, not compiled, not run on hardware
//
#include <hip/hip_runtime.h>
#include <hip/hip_bf16.h>

// ---------------------------------------------------------------------------
// Problem constants (match the reference)
// ---------------------------------------------------------------------------
#define N_TOK 4096
#define D_DIM 2048
#define F_DIM 16384
#define K_WIN 32

typedef __attribute__((ext_vector_type(16))) __bf16    v16bf;
typedef __attribute__((ext_vector_type(8)))  float     v8f;
typedef __attribute__((ext_vector_type(4)))  unsigned  v4u;
typedef __attribute__((ext_vector_type(8)))  unsigned  v8u;
typedef __attribute__((ext_vector_type(4)))  unsigned  u32x4;
typedef __attribute__((ext_vector_type(8)))  int       i32x8;
typedef __attribute__((ext_vector_type(4)))  int       i32x4;

static __device__ inline v16bf make_frag(v4u lo, v4u hi) {
    v8u u;
    u[0] = lo[0]; u[1] = lo[1]; u[2] = lo[2]; u[3] = lo[3];
    u[4] = hi[0]; u[5] = hi[1]; u[6] = hi[2]; u[7] = hi[3];
    return __builtin_bit_cast(v16bf, u);
}

// ---------------------------------------------------------------------------
// TDM: async-load one 2D bf16 tile (tile_d1 rows x tile_d0 elems) from global
// to LDS with row padding of 16B after every 128B of row data (matches the
// LDT = BK+8 padded LDS layout). Tracked by TENSORcnt.
//   D# group0: count=1 | lds_addr | global_addr(57b) | type=2
//   D# group1: data_size=2B, pad_enable, pad_interval=32dw(code 4),
//              pad_amount=4dw(code 3), tensor dims, tile dims, dim0 stride
// This toolchain's builtin takes 6 args (g0, g1, g2, g3, g4, cpol).
// ---------------------------------------------------------------------------
static __device__ inline void tdm_load_tile_bf16(
    unsigned lds_addr, const void* gptr,
    unsigned tile_d0, unsigned tile_d1,
    unsigned tensor_d0, unsigned tensor_d1, unsigned stride_elems)
{
    unsigned long long ga = (unsigned long long)(uintptr_t)gptr;
    u32x4 g0;
    g0[0] = 1u;                                              // count=1 (valid)
    g0[1] = lds_addr;                                        // LDS byte address
    g0[2] = (unsigned)(ga & 0xFFFFFFFFu);                    // global_addr[31:0]
    g0[3] = ((unsigned)(ga >> 32) & 0x01FFFFFFu) | 0x80000000u; // [56:32] | type=2

    i32x8 g1;
    g1[0] = (int)((1u << 16)      // data_size = 1 -> 2 bytes
                | (1u << 20)      // pad_enable
                | (4u << 22)      // pad_interval code 4 -> every 32 DWORDs
                | (3u << 25));    // pad_amount  code 3 -> 4 DWORDs (16 B)
    g1[1] = (int)((tensor_d0 & 0xFFFFu) << 16);              // tensor_dim0[15:0]
    g1[2] = (int)(((tensor_d0 >> 16) & 0xFFFFu)
                | ((tensor_d1 & 0xFFFFu) << 16));            // dim0 hi | dim1 lo
    g1[3] = (int)(((tensor_d1 >> 16) & 0xFFFFu)
                | ((tile_d0 & 0xFFFFu) << 16));              // dim1 hi | tile_dim0
    g1[4] = (int)(tile_d1 & 0xFFFFu);                        // tile_dim1 (tile_dim2=0)
    g1[5] = (int)stride_elems;                               // tensor_dim0_stride[31:0]
    g1[6] = 0;                                               // stride hi, dim1_stride lo
    g1[7] = 0;

    i32x4 gz4 = {0, 0, 0, 0};
    i32x8 gz8 = {0, 0, 0, 0, 0, 0, 0, 0};
    __builtin_amdgcn_tensor_load_to_lds(g0, g1, gz4, gz4, gz8, 0);
}

// ---------------------------------------------------------------------------
// Kernel 1: fp32 -> bf16 (round-to-nearest-even) for x and W; zero the loss.
// ---------------------------------------------------------------------------
__global__ __launch_bounds__(256) void sae_convert(
    const float* __restrict__ x, const float* __restrict__ W,
    unsigned short* __restrict__ xb, unsigned short* __restrict__ wb,
    float* __restrict__ loss)
{
    size_t i      = (size_t)blockIdx.x * blockDim.x + threadIdx.x;
    size_t stride = (size_t)gridDim.x * blockDim.x;
    if (i == 0) *loss = 0.0f;
    const size_t nx = (size_t)N_TOK * D_DIM;
    const size_t nw = (size_t)F_DIM * D_DIM;
    for (; i < nx + nw; i += stride) {
        float v;
        if (i < nx) v = x[i]; else v = W[i - nx];
        unsigned u = __builtin_bit_cast(unsigned, v);
        unsigned short b = (unsigned short)((u + 0x7FFFu + ((u >> 16) & 1u)) >> 16);
        if (i < nx) xb[i] = b; else wb[i - nx] = b;
    }
}

// ---------------------------------------------------------------------------
// Kernel 2: raw = x (N_TOK x D_DIM) * W^T, bf16 WMMA with f32 accumulation.
// 128x128 block tile, BK=64, 8 waves (4M x 2N), 8 v_wmma per wave per k-step.
// Staging: TDM tensor_load_to_lds, double-buffered LDS, software pipelined.
// ---------------------------------------------------------------------------
#define BM 128
#define BN 128
#define BK 64
#define LDT (BK + 8)   // padded row; 144 B stride, produced by TDM pad feature

__global__ __launch_bounds__(256) void sae_encode_gemm(
    const unsigned short* __restrict__ xb,   // [N_TOK][D_DIM] bf16
    const unsigned short* __restrict__ wb,   // [F_DIM][D_DIM] bf16
    float* __restrict__ raw)                 // [N_TOK][F_DIM]
{
    __shared__ unsigned short As[2][BM * LDT];  // 2 x 18 KB
    __shared__ unsigned short Bs[2][BN * LDT];  // 2 x 18 KB

    const int tid   = threadIdx.x;
    const int lane  = tid & 31;
    const int wave  = tid >> 5;
    const int waveM = wave & 3;      // 0..3  -> M offset waveM*32
    const int waveN = wave >> 2;     // 0..1  -> N offset waveN*64
    const int blockM = blockIdx.y * BM;
    const int blockF = blockIdx.x * BN;

    const v8f vzero = {0.f,0.f,0.f,0.f,0.f,0.f,0.f,0.f};
    v8f acc[2][4];
#pragma unroll
    for (int mt = 0; mt < 2; ++mt)
#pragma unroll
        for (int nt = 0; nt < 4; ++nt) acc[mt][nt] = vzero;

    const int l15 = lane & 15;
    const int hiHalf = (lane & 16) ? 1 : 0;

    const unsigned short* gA = xb + (size_t)blockM * D_DIM;
    const unsigned short* gB = wb + (size_t)blockF * D_DIM;
    const int steps = D_DIM / BK;    // 32

    // Prologue: kick off DMA of the first tile pair.
    if (tid < 32) {
        tdm_load_tile_bf16((unsigned)(uintptr_t)&As[0][0], gA, BK, BM, D_DIM, BM, D_DIM);
        tdm_load_tile_bf16((unsigned)(uintptr_t)&Bs[0][0], gB, BK, BN, D_DIM, BN, D_DIM);
    }

    for (int i = 0; i < steps; ++i) {
        if (tid < 32) __builtin_amdgcn_s_wait_tensorcnt(0);
        __syncthreads();   // tile i visible to all waves; buffer i+1 free of readers

        if (tid < 32 && (i + 1) < steps) {   // DMA next tiles while computing
            int b = (i + 1) & 1;
            tdm_load_tile_bf16((unsigned)(uintptr_t)&As[b][0], gA + (size_t)(i + 1) * BK,
                               BK, BM, D_DIM, BM, D_DIM);
            tdm_load_tile_bf16((unsigned)(uintptr_t)&Bs[b][0], gB + (size_t)(i + 1) * BK,
                               BK, BN, D_DIM, BN, D_DIM);
        }

        const unsigned short* sA = As[i & 1];
        const unsigned short* sB = Bs[i & 1];

#pragma unroll
        for (int ks = 0; ks < BK; ks += 32) {
            // A frags: lanes 0-15 hold K[0..7],K[16..23]; lanes 16-31 K[8..15],K[24..31]
            v16bf afrag[2];
            const int pa = hiHalf ? 8 : 0;
#pragma unroll
            for (int mt = 0; mt < 2; ++mt) {
                int row  = waveM * 32 + mt * 16 + l15;
                int base = row * LDT + ks + pa;
                v4u lo = *reinterpret_cast<const v4u*>(&sA[base]);
                v4u hi = *reinterpret_cast<const v4u*>(&sA[base + 16]);
                afrag[mt] = make_frag(lo, hi);
            }
            // B frags: lanes 0-15 (col n) hold K[0..7],K[16..23]; lanes 16-31 K[8..15],K[24..31]
            v16bf bfrag[4];
            const int pb = hiHalf ? 8 : 0;
#pragma unroll
            for (int nt = 0; nt < 4; ++nt) {
                int row  = waveN * 64 + nt * 16 + l15;
                int base = row * LDT + ks + pb;
                v4u lo = *reinterpret_cast<const v4u*>(&sB[base]);
                v4u hi = *reinterpret_cast<const v4u*>(&sB[base + 16]);
                bfrag[nt] = make_frag(lo, hi);
            }
#pragma unroll
            for (int mt = 0; mt < 2; ++mt)
#pragma unroll
                for (int nt = 0; nt < 4; ++nt)
                    acc[mt][nt] = __builtin_amdgcn_wmma_f32_16x16x32_bf16(
                        false, afrag[mt], false, bfrag[nt],
                        (short)0, acc[mt][nt], false, false);
        }
        __syncthreads();   // all reads of buffer i done before its refill next+1
    }

    // ---- write back: C/D layout: VGPR r, lanes0-15 -> M=r, lanes16-31 -> M=r+8
    const int rowOff = hiHalf ? 8 : 0;
#pragma unroll
    for (int mt = 0; mt < 2; ++mt) {
#pragma unroll
        for (int nt = 0; nt < 4; ++nt) {
            int fcol = blockF + waveN * 64 + nt * 16 + l15;
#pragma unroll
            for (int r = 0; r < 8; ++r) {
                int nrow = blockM + waveM * 32 + mt * 16 + r + rowOff;
                raw[(size_t)nrow * F_DIM + fcol] = acc[mt][nt][r];
            }
        }
    }
}

// ---------------------------------------------------------------------------
// Kernel 3: per-row top-32 of 16384 scores; mask acts in place; emit (idx,val).
// One workgroup (256 threads) per row; whole row lives in LDS (64 KB).
// ---------------------------------------------------------------------------
__global__ __launch_bounds__(256) void sae_topk_mask(
    float* __restrict__ acts,            // in: raw scores, out: masked acts
    int* __restrict__ widx, float* __restrict__ wval)
{
    __shared__ float    srow[F_DIM];       // 64 KB
    __shared__ unsigned sflag[F_DIM / 32]; // 2 KB winner bitmask
    __shared__ float    sval[256];
    __shared__ int      sidx[256];

    const int n   = blockIdx.x;
    const int tid = threadIdx.x;
    float* rowp = acts + (size_t)n * F_DIM;

    for (int i = tid; i < F_DIM; i += 256) srow[i] = rowp[i];
    for (int i = tid; i < F_DIM / 32; i += 256) sflag[i] = 0u;
    __syncthreads();

    for (int r = 0; r < K_WIN; ++r) {
        float best = -1e30f;
        int   bi   = 0;
        for (int i = tid; i < F_DIM; i += 256) {
            if (!((sflag[i >> 5] >> (i & 31)) & 1u)) {
                float v = srow[i];
                if (v > best) { best = v; bi = i; }
            }
        }
        sval[tid] = best; sidx[tid] = bi;
        __syncthreads();
        for (int s = 128; s > 0; s >>= 1) {
            if (tid < s) {
                if (sval[tid + s] > sval[tid]) { sval[tid] = sval[tid + s]; sidx[tid] = sidx[tid + s]; }
            }
            __syncthreads();
        }
        if (tid == 0) {
            int w = sidx[0];
            sflag[w >> 5] |= (1u << (w & 31));
            widx[n * K_WIN + r] = w;
            wval[n * K_WIN + r] = sval[0];
        }
        __syncthreads();
    }

    for (int i = tid; i < F_DIM; i += 256) {
        bool sel = (sflag[i >> 5] >> (i & 31)) & 1u;
        rowp[i] = sel ? srow[i] : 0.0f;
    }
}

// ---------------------------------------------------------------------------
// Kernel 4: recon[n] = sum_j val[n][j] * W[idx[n][j]]; loss += ||recon-x||^2/N.
// One workgroup per row; W is fp32 and L2-resident (134 MB < 192 MB L2).
// ---------------------------------------------------------------------------
__global__ __launch_bounds__(256) void sae_decode_loss(
    const float* __restrict__ W, const float* __restrict__ x,
    const int* __restrict__ widx, const float* __restrict__ wval,
    float* __restrict__ recon, float* __restrict__ loss)
{
    __shared__ int   sidx[K_WIN];
    __shared__ float sval[K_WIN];
    __shared__ float sred[256];

    const int n   = blockIdx.x;
    const int tid = threadIdx.x;
    if (tid < K_WIN) { sidx[tid] = widx[n * K_WIN + tid]; sval[tid] = wval[n * K_WIN + tid]; }
    __syncthreads();

    float acc[8] = {0.f,0.f,0.f,0.f,0.f,0.f,0.f,0.f};
#pragma unroll 4
    for (int j = 0; j < K_WIN; ++j) {
        const float* wr = W + (size_t)sidx[j] * D_DIM;
        float a = sval[j];
#pragma unroll
        for (int c = 0; c < 8; ++c) acc[c] += a * wr[tid + c * 256];
    }

    const float* xr = x + (size_t)n * D_DIM;
    float*       rr = recon + (size_t)n * D_DIM;
    float ssq = 0.f;
#pragma unroll
    for (int c = 0; c < 8; ++c) {
        float v = acc[c];
        rr[tid + c * 256] = v;
        float d = v - xr[tid + c * 256];
        ssq += d * d;
    }
    sred[tid] = ssq;
    __syncthreads();
    for (int s = 128; s > 0; s >>= 1) {
        if (tid < s) sred[tid] += sred[tid + s];
        __syncthreads();
    }
    if (tid == 0) atomicAdd(loss, sred[0] * (1.0f / (float)N_TOK));
}

// ---------------------------------------------------------------------------
// Host launcher
// ---------------------------------------------------------------------------
extern "C" void kernel_launch(void* const* d_in, const int* in_sizes, int n_in,
                              void* d_out, int out_size, void* d_ws, size_t ws_size,
                              hipStream_t stream) {
    const float* x = (const float*)d_in[0];   // [N_TOK * D_DIM]
    const float* W = (const float*)d_in[1];   // [F_DIM * D_DIM]

    float* loss  = (float*)d_out;                       // [1]
    float* recon = loss + 1;                            // [N_TOK * D_DIM]
    float* acts  = recon + (size_t)N_TOK * D_DIM;       // [N_TOK * F_DIM]

    char* ws = (char*)d_ws;
    const size_t xb_bytes = (size_t)N_TOK * D_DIM * sizeof(unsigned short);  // 16 MB
    const size_t wb_bytes = (size_t)F_DIM * D_DIM * sizeof(unsigned short);  // 64 MB
    const size_t ix_bytes = (size_t)N_TOK * K_WIN * sizeof(int);             // 512 KB
    unsigned short* xb   = (unsigned short*)ws;
    unsigned short* wb   = (unsigned short*)(ws + xb_bytes);
    int*            widx = (int*)(ws + xb_bytes + wb_bytes);
    float*          wval = (float*)(ws + xb_bytes + wb_bytes + ix_bytes);

    sae_convert<<<2048, 256, 0, stream>>>(x, W, xb, wb, loss);

    dim3 ggrid(F_DIM / BN, N_TOK / BM);   // 128 x 32
    sae_encode_gemm<<<ggrid, 256, 0, stream>>>(xb, wb, acts);

    sae_topk_mask<<<N_TOK, 256, 0, stream>>>(acts, widx, wval);

    sae_decode_loss<<<N_TOK, 256, 0, stream>>>(W, x, widx, wval, recon, loss);
}
